// ResidualVectorQuantizer_64476049047581
// MI455X (gfx1250) — matrix operand order repaired and checked by reading, not compiled, not run on hardware
//
#include <hip/hip_runtime.h>
#include <hip/hip_bf16.h>
#include <math.h>

typedef __attribute__((ext_vector_type(16))) __bf16 v16bf;
typedef __attribute__((ext_vector_type(8)))  __bf16 v8bf;
typedef __attribute__((ext_vector_type(8)))  float  v8f;

#define N_CB   10
#define KCODES 2048
#define DIM    256
#define SEMD   256
#define W2VD   1024
#define MTOK   16384                 // B*T = 16*1024
#define NELEM  (MTOK * DIM)          // 4,194,304

// d_out layout (floats): z_q_total | codes (MTOK x N_CB) | vq_loss | semantic_loss
#define OFF_CODES   (MTOK * DIM)
#define OFF_VQLOSS  (OFF_CODES + MTOK * N_CB)
#define OFF_SEMLOSS (OFF_VQLOSS + 1)

// ---------------------------------------------------------------------------
// init: residual = z (f32 + bf16 mirror); zero loss accumulators
// ---------------------------------------------------------------------------
__global__ void rvq_init_kernel(const float* __restrict__ z,
                                float* __restrict__ residual,
                                __bf16* __restrict__ residual_bf,
                                float* __restrict__ accum) {
  size_t i = (size_t)blockIdx.x * blockDim.x + threadIdx.x;
  size_t stride = (size_t)gridDim.x * blockDim.x;
  for (size_t k = i; k < (size_t)NELEM; k += stride) {
    const float v = z[k];
    residual[k] = v;
    residual_bf[k] = (__bf16)v;
  }
  if (i == 0) { accum[0] = 0.0f; accum[1] = 0.0f; }
}

// ---------------------------------------------------------------------------
// convert full codebook tensor to bf16 (row-major, same layout)
// ---------------------------------------------------------------------------
__global__ void rvq_cvt_bf16_kernel(const float* __restrict__ src,
                                    __bf16* __restrict__ dst, int n) {
  size_t i = (size_t)blockIdx.x * blockDim.x + threadIdx.x;
  size_t stride = (size_t)gridDim.x * blockDim.x;
  for (size_t k = i; k < (size_t)n; k += stride) dst[k] = (__bf16)src[k];
}

// ---------------------------------------------------------------------------
// transpose (K,N) f32 -> (N,K) bf16, so WMMA B-operand reads are contiguous
// ---------------------------------------------------------------------------
__global__ void rvq_transpose_bf16_kernel(const float* __restrict__ src,
                                          __bf16* __restrict__ dst,
                                          int K, int N) {
  int i = blockIdx.x * blockDim.x + threadIdx.x;
  if (i >= K * N) return;
  const int n = i / K, k = i % K;
  dst[(size_t)n * K + k] = (__bf16)src[(size_t)k * N + n];
}

// ---------------------------------------------------------------------------
// halfnorm[c] = 0.5 * ||E[c]||^2  for all N_CB*KCODES codes
// ---------------------------------------------------------------------------
__global__ void rvq_halfnorm_kernel(const float* __restrict__ E,
                                    float* __restrict__ halfnorm) {
  int c = blockIdx.x * blockDim.x + threadIdx.x;
  if (c >= N_CB * KCODES) return;
  const float* row = E + (size_t)c * DIM;
  float s = 0.0f;
#pragma unroll 4
  for (int d = 0; d < DIM; ++d) { float v = row[d]; s += v * v; }
  halfnorm[c] = 0.5f * s;
}

// ---------------------------------------------------------------------------
// helper: build the 16-bit WMMA A-fragment for one K-step from a bf16 row.
// lane holds row; elements 0..15 -> K = kb..kb+15
// ---------------------------------------------------------------------------
__device__ __forceinline__ v16bf load_a_frag(const __bf16* __restrict__ row, int kb) {
  v8bf lo = *(const v8bf*)(row + kb);
  v8bf hi = *(const v8bf*)(row + kb + 8);
  return __builtin_shufflevector(lo, hi, 0, 1, 2, 3, 4, 5, 6, 7,
                                 8, 9, 10, 11, 12, 13, 14, 15);
}

// ---------------------------------------------------------------------------
// Fused GEMM + row-argmax: idx[m] = argmax_k ( residual[m]·E[k] - 0.5||E[k]||^2 )
// One wave per 16-row M tile, 8 waves/block, grid = MTOK/128.
// Register double-buffering: B fragments of tile nt+1 load during tile nt WMMAs.
// ---------------------------------------------------------------------------
__global__ void __launch_bounds__(256)
rvq_argmin_kernel(const __bf16* __restrict__ residual_bf,
                  const __bf16* __restrict__ Ebf,    // bf16 E + cb*KCODES*DIM
                  const float* __restrict__ hn,      // halfnorm + cb*KCODES
                  int* __restrict__ idx_out,
                  float* __restrict__ codes_out)     // d_out + OFF_CODES + cb
{
  const int lane = threadIdx.x & 31;
  const int wave = threadIdx.x >> 5;
  const int half = lane >> 4;        // which half-wave
  const int lrow = lane & 15;        // A row / B,C column within tile
  const int m0 = (blockIdx.x * 8 + wave) * 16;

  // Preload A tile (16x256) as WMMA A fragments in contiguous k order (K = ks*32 + 16*half + s)
  v16bf a[8];
  const __bf16* arow = residual_bf + (size_t)(m0 + lrow) * DIM;
#pragma unroll
  for (int ks = 0; ks < 8; ++ks) a[ks] = load_a_frag(arow, ks * 32 + half * 16);

  float bestv[8];
  int   besti[8];
#pragma unroll
  for (int v = 0; v < 8; ++v) { bestv[v] = -3.0e38f; besti[v] = 0; }

  // B layout: lane holds column n0+lrow; element j -> K = ks*32 + half*16 + j
  const __bf16* erow = Ebf + (size_t)lrow * DIM + half * 16;
  const size_t tile_step = (size_t)16 * DIM;        // 16 codes per N tile

  v16bf b0[8], b1[8];
#pragma unroll
  for (int ks = 0; ks < 8; ++ks) b0[ks] = *(const v16bf*)(erow + ks * 32);

  for (int nt = 0; nt < KCODES / 16; nt += 2) {
    const __bf16* ep1 = erow + (size_t)(nt + 1) * tile_step;
    __builtin_prefetch(ep1 + tile_step, 0, 1);       // prefetch tile nt+2
    const float h0 = hn[nt * 16 + lrow];
    const float h1 = hn[nt * 16 + 16 + lrow];

    // preload tile nt+1 while computing tile nt
#pragma unroll
    for (int ks = 0; ks < 8; ++ks) b1[ks] = *(const v16bf*)(ep1 + ks * 32);

    v8f c = {};
#pragma unroll
    for (int ks = 0; ks < 8; ++ks)
      c = __builtin_amdgcn_wmma_f32_16x16x32_bf16(false, a[ks], false, b0[ks],
                                                  (short)0, c, false, false);
    {
      const int vid = nt * 16 + lrow;
#pragma unroll
      for (int v = 0; v < 8; ++v) {
        const float val = c[v] - h0;     // strict > keeps first (lowest) index
        if (val > bestv[v]) { bestv[v] = val; besti[v] = vid; }
      }
    }

    // preload tile nt+2 while computing tile nt+1
    if (nt + 2 < KCODES / 16) {
      const __bf16* ep2 = erow + (size_t)(nt + 2) * tile_step;
#pragma unroll
      for (int ks = 0; ks < 8; ++ks) b0[ks] = *(const v16bf*)(ep2 + ks * 32);
    }

    c = (v8f){};
#pragma unroll
    for (int ks = 0; ks < 8; ++ks)
      c = __builtin_amdgcn_wmma_f32_16x16x32_bf16(false, a[ks], false, b1[ks],
                                                  (short)0, c, false, false);
    {
      const int vid = nt * 16 + 16 + lrow;
#pragma unroll
      for (int v = 0; v < 8; ++v) {
        const float val = c[v] - h1;
        if (val > bestv[v]) { bestv[v] = val; besti[v] = vid; }
      }
    }
  }

  // single cross-lane argmax butterfly at the end (16-lane segments);
  // index tie-break here preserves argmin's first-hit semantics across lanes
#pragma unroll
  for (int v = 0; v < 8; ++v) {
    float val = bestv[v];
    int   vid = besti[v];
#pragma unroll
    for (int off = 1; off < 16; off <<= 1) {
      float ov = __shfl_xor(val, off, 16);
      int   oi = __shfl_xor(vid, off, 16);
      if (ov > val || (ov == val && oi < vid)) { val = ov; vid = oi; }
    }
    bestv[v] = val; besti[v] = vid;
  }

  if (lrow == 0) {   // lane 0 -> rows m0..m0+7, lane 16 -> rows m0+8..m0+15
#pragma unroll
    for (int v = 0; v < 8; ++v) {
      const int m = m0 + half * 8 + v;
      idx_out[m] = besti[v];
      codes_out[(size_t)m * N_CB] = (float)besti[v];
    }
  }
}

// ---------------------------------------------------------------------------
// update: q = E[idx]; residual -= q (f32 + bf16 mirror); accum += sum(r^2).
// z_q_total is NOT accumulated here (it equals z - residual_final).
// ---------------------------------------------------------------------------
__global__ void __launch_bounds__(256)
rvq_update_kernel(float* __restrict__ residual,
                  __bf16* __restrict__ residual_bf,
                  const float* __restrict__ Ecb,
                  const int* __restrict__ idx,
                  float* __restrict__ accum)
{
  __shared__ float wsum[8];
  float local = 0.0f;
  size_t i = (size_t)blockIdx.x * blockDim.x + threadIdx.x;
  size_t stride = (size_t)gridDim.x * blockDim.x;
  for (size_t k = i; k < (size_t)NELEM; k += stride) {
    const int n = (int)(k >> 8);
    const int d = (int)(k & 255);
    const float q = Ecb[(size_t)idx[n] * DIM + d];
    const float r = residual[k] - q;
    residual[k] = r;
    residual_bf[k] = (__bf16)r;
    local += r * r;
  }
#pragma unroll
  for (int off = 16; off >= 1; off >>= 1) local += __shfl_xor(local, off, 32);
  const int lane = threadIdx.x & 31, wv = threadIdx.x >> 5;
  if (lane == 0) wsum[wv] = local;
  __syncthreads();
  if (threadIdx.x == 0) {
    float s = 0.0f;
    for (int w = 0; w < 8; ++w) s += wsum[w];
    atomicAdd(accum, s);
  }
}

// ---------------------------------------------------------------------------
// z_q_total = z - residual_final  (one pass at the end)
// ---------------------------------------------------------------------------
__global__ void rvq_zq_kernel(const float* __restrict__ z,
                              const float* __restrict__ residual,
                              float* __restrict__ zq) {
  size_t i = (size_t)blockIdx.x * blockDim.x + threadIdx.x;
  size_t stride = (size_t)gridDim.x * blockDim.x;
  for (size_t k = i; k < (size_t)NELEM; k += stride) zq[k] = z[k] - residual[k];
}

// ---------------------------------------------------------------------------
// semantic GEMM 1: H = gelu_exact( gather(E0, idx0) @ W1 + b1 ), H stored bf16
// ---------------------------------------------------------------------------
__global__ void __launch_bounds__(256)
rvq_semantic1_kernel(const __bf16* __restrict__ E0bf,
                     const int* __restrict__ idx,
                     const __bf16* __restrict__ W1t, // (SEMD, DIM) bf16, transposed
                     const float* __restrict__ b1,
                     __bf16* __restrict__ Hbf)
{
  const int lane = threadIdx.x & 31;
  const int wave = threadIdx.x >> 5;
  const int half = lane >> 4;
  const int lrow = lane & 15;
  const int m0 = (blockIdx.x * 8 + wave) * 16;

  v16bf a[8];
  const __bf16* arow = E0bf + (size_t)idx[m0 + lrow] * DIM;   // q0 row gather
#pragma unroll
  for (int ks = 0; ks < 8; ++ks) a[ks] = load_a_frag(arow, ks * 32 + half * 16);

  for (int nt = 0; nt < SEMD / 16; ++nt) {
    const int n = nt * 16 + lrow;
    const __bf16* wrow = W1t + (size_t)n * DIM + half * 16;
    v8f c = {};
#pragma unroll
    for (int ks = 0; ks < 8; ++ks) {
      const v16bf b = *(const v16bf*)(wrow + ks * 32);
      c = __builtin_amdgcn_wmma_f32_16x16x32_bf16(false, a[ks], false, b,
                                                  (short)0, c, false, false);
    }
    const float bias = b1[n];
#pragma unroll
    for (int v = 0; v < 8; ++v) {
      const float x = c[v] + bias;
      const float g = 0.5f * x * (1.0f + erff(x * 0.7071067811865475f));
      Hbf[(size_t)(m0 + half * 8 + v) * SEMD + n] = (__bf16)g;
    }
  }
}

// ---------------------------------------------------------------------------
// semantic GEMM 2 + fused MSE: accum += sum((H @ W2 + b2 - w2v)^2)
// ---------------------------------------------------------------------------
__global__ void __launch_bounds__(256)
rvq_semantic2_kernel(const __bf16* __restrict__ Hbf,
                     const __bf16* __restrict__ W2t, // (W2VD, SEMD) bf16, transposed
                     const float* __restrict__ b2,
                     const float* __restrict__ w2v,
                     float* __restrict__ accum_sem)
{
  __shared__ float wsum[8];
  const int lane = threadIdx.x & 31;
  const int wave = threadIdx.x >> 5;
  const int half = lane >> 4;
  const int lrow = lane & 15;
  const int m0 = (blockIdx.x * 8 + wave) * 16;

  v16bf a[8];
  const __bf16* arow = Hbf + (size_t)(m0 + lrow) * SEMD;
#pragma unroll
  for (int ks = 0; ks < 8; ++ks) a[ks] = load_a_frag(arow, ks * 32 + half * 16);

  float local = 0.0f;
  for (int nt = 0; nt < W2VD / 16; ++nt) {
    const int n = nt * 16 + lrow;
    const __bf16* wrow = W2t + (size_t)n * SEMD + half * 16;
    v8f c = {};
#pragma unroll
    for (int ks = 0; ks < 8; ++ks) {
      const v16bf b = *(const v16bf*)(wrow + ks * 32);
      c = __builtin_amdgcn_wmma_f32_16x16x32_bf16(false, a[ks], false, b,
                                                  (short)0, c, false, false);
    }
    const float bias = b2[n];
#pragma unroll
    for (int v = 0; v < 8; ++v) {
      const int m = m0 + half * 8 + v;
      const float diff = c[v] + bias - w2v[(size_t)m * W2VD + n];
      local += diff * diff;
    }
  }
#pragma unroll
  for (int off = 16; off >= 1; off >>= 1) local += __shfl_xor(local, off, 32);
  if (lane == 0) wsum[wave] = local;
  __syncthreads();
  if (threadIdx.x == 0) {
    float s = 0.0f;
    for (int w = 0; w < 8; ++w) s += wsum[w];
    atomicAdd(accum_sem, s);
  }
}

// ---------------------------------------------------------------------------
// finalize: scale accumulators into d_out scalars
// ---------------------------------------------------------------------------
__global__ void rvq_finalize_kernel(const float* __restrict__ accum,
                                    float* __restrict__ out) {
  if (blockIdx.x == 0 && threadIdx.x == 0) {
    out[OFF_VQLOSS]  = 1.25f * accum[0] / (float)NELEM;          // (1+COMMIT)*mean
    out[OFF_SEMLOSS] = accum[1] / ((float)MTOK * (float)W2VD);
  }
}

// ---------------------------------------------------------------------------
extern "C" void kernel_launch(void* const* d_in, const int* in_sizes, int n_in,
                              void* d_out, int out_size, void* d_ws, size_t ws_size,
                              hipStream_t stream) {
  (void)in_sizes; (void)n_in; (void)out_size; (void)ws_size;
  const float* z   = (const float*)d_in[0];
  const float* w2v = (const float*)d_in[1];
  const float* E   = (const float*)d_in[2];
  const float* W1  = (const float*)d_in[3];
  const float* b1  = (const float*)d_in[4];
  const float* W2  = (const float*)d_in[5];
  const float* b2  = (const float*)d_in[6];
  float* out = (float*)d_out;

  // workspace layout (bytes)
  char* ws = (char*)d_ws;
  size_t off = 0;
  float*  residual    = (float*)(ws + off);  off += (size_t)NELEM * 4;        // 16 MB
  __bf16* residual_bf = (__bf16*)(ws + off); off += (size_t)NELEM * 2;        //  8 MB
  __bf16* Hbf         = (__bf16*)(ws + off); off += (size_t)MTOK * SEMD * 2;  //  8 MB
  __bf16* Ebf         = (__bf16*)(ws + off); off += (size_t)N_CB * KCODES * DIM * 2; // 10 MB
  __bf16* W1t         = (__bf16*)(ws + off); off += (size_t)SEMD * DIM * 2;   // 128 KB
  __bf16* W2t         = (__bf16*)(ws + off); off += (size_t)W2VD * SEMD * 2;  // 512 KB
  float*  halfnorm    = (float*)(ws + off);  off += (size_t)N_CB * KCODES * 4;
  int*    idx         = (int*)(ws + off);    off += (size_t)MTOK * 4;
  float*  accum       = (float*)(ws + off);

  rvq_init_kernel<<<512, 256, 0, stream>>>(z, residual, residual_bf, accum);
  rvq_cvt_bf16_kernel<<<2048, 256, 0, stream>>>(E, Ebf, N_CB * KCODES * DIM);
  rvq_halfnorm_kernel<<<(N_CB * KCODES + 255) / 256, 256, 0, stream>>>(E, halfnorm);
  rvq_transpose_bf16_kernel<<<(DIM * SEMD + 255) / 256, 256, 0, stream>>>(W1, W1t, DIM, SEMD);
  rvq_transpose_bf16_kernel<<<(SEMD * W2VD + 255) / 256, 256, 0, stream>>>(W2, W2t, SEMD, W2VD);

  for (int cb = 0; cb < N_CB; ++cb) {
    rvq_argmin_kernel<<<MTOK / 128, 256, 0, stream>>>(
        residual_bf, Ebf + (size_t)cb * KCODES * DIM, halfnorm + cb * KCODES,
        idx, out + OFF_CODES + cb);
    rvq_update_kernel<<<1024, 256, 0, stream>>>(
        residual, residual_bf, E + (size_t)cb * KCODES * DIM, idx, accum);
    if (cb == 0) {
      // idx still holds codebook-0 indices here
      rvq_semantic1_kernel<<<MTOK / 128, 256, 0, stream>>>(Ebf, idx, W1t, b1, Hbf);
      rvq_semantic2_kernel<<<MTOK / 128, 256, 0, stream>>>(Hbf, W2t, b2, w2v, accum + 1);
    }
  }
  rvq_zq_kernel<<<512, 256, 0, stream>>>(z, residual, out);
  rvq_finalize_kernel<<<1, 64, 0, stream>>>(accum, out);
}
